// BatchedCrossAttentionXSMM_52948356825341
// MI455X (gfx1250) — matrix operand order repaired and checked by reading, not compiled, not run on hardware
//
#include <hip/hip_runtime.h>
#include <hip/hip_bf16.h>
#include <math.h>

// ---------------------------------------------------------------------------
// Fused gated cross-attention for MI455X (gfx1250), wave32 + WMMA bf16.
//   B=1, NQ=NK=2048, A_DIM=M_DIM=1024, H=16, KD=VD=64
// prep -> 4x WMMA projection GEMMs -> flash attention (row-per-lane softmax)
// All per-chunk operands are preloaded into registers ahead of the WMMAs so
// the scheduler can clause the loads and use partial s_wait_loadcnt waits.
// ---------------------------------------------------------------------------

typedef __bf16 bf16_t;
typedef __attribute__((ext_vector_type(16))) __bf16 v16bf;
typedef __attribute__((ext_vector_type(8)))  __bf16 v8bf;
typedef __attribute__((ext_vector_type(8)))  float  v8f;

#define NQS 2048
#define NKS 2048
#define ADIM 1024      // = H*KD = H*VD
#define NHEAD 16
#define HDIM 64
#define SROW 40        // padded LDS row stride (floats): 32B-aligned, conflict-reduced

__device__ __forceinline__ v8f wmma_bf16(v16bf a, v16bf b, v8f c) {
  // v_wmma_f32_16x16x32_bf16: D = A(16x32) * B(32x16) + C(16x16 f32)
  return __builtin_amdgcn_wmma_f32_16x16x32_bf16(false, a, false, b, (short)0, c,
                                                 false, false);
}

union AFrag { v16bf v; v8bf h2[2]; __bf16 e[16]; };

// ---------------------------------------------------------------------------
// prep kernels
// ---------------------------------------------------------------------------
__global__ void cvt_f32_to_bf16(const float* __restrict__ in,
                                bf16_t* __restrict__ out, int n) {
  int i = blockIdx.x * blockDim.x + threadIdx.x;
  if (i < n) out[i] = (bf16_t)in[i];
}

// W[k][n] (f32, 1024x1024) -> Wt[n][k] (bf16), N-major as required by B-fragments
__global__ void transpose_to_bf16(const float* __restrict__ in,
                                  bf16_t* __restrict__ out) {
  int k = blockIdx.x * 16 + threadIdx.x;
  int n = blockIdx.y * 16 + threadIdx.y;
  out[(size_t)n * ADIM + k] = (bf16_t)in[(size_t)k * ADIM + n];
}

// ---------------------------------------------------------------------------
// Projection GEMM: C[M x 1024] = A[M x 1024] * W[1024 x 1024]
// 128 threads = 4 waves; block tile 64(M) x 64(N); wave = 16x64 strip.
// Fragments straight from global (operands fully resident in 192MB L2).
// MODE: 0=(x+bias)*scale->bf16  1=x->bf16  2=x->bf16 V^T store  3=sigmoid->f32
// ---------------------------------------------------------------------------
template <int MODE>
__global__ __launch_bounds__(128) void proj_gemm(
    const bf16_t* __restrict__ A, const bf16_t* __restrict__ Wt,
    const float* __restrict__ bias, bf16_t* __restrict__ outb,
    float* __restrict__ outf, float scale) {
  const int lane = threadIdx.x & 31;
  const int wave = threadIdx.x >> 5;
  const int mrow = lane & 15;   // A-frag M row / C-frag N col (ISA layouts)
  const int g    = lane >> 4;   // lane half-group
  const int m0   = blockIdx.x * 64 + wave * 16;
  const int n0   = blockIdx.y * 64;

  v8f acc[4] = {};
  const bf16_t* arow = A + (size_t)(m0 + mrow) * ADIM;

#pragma unroll 2
  for (int k = 0; k < ADIM; k += 32) {
    // Preload A-frag + all 4 B-frags, then issue the 4 WMMAs.
    AFrag a;  // A-frag 16x32: lane group g holds K = g*16..+15
    a.h2[0] = *(const v8bf*)(arow + k + g * 16);
    a.h2[1] = *(const v8bf*)(arow + k + g * 16 + 8);
    v16bf b[4];
#pragma unroll
    for (int f = 0; f < 4; ++f)  // B-frag 32x16: lane n holds K = k+g*16..+15
      b[f] = *(const v16bf*)(Wt + (size_t)(n0 + f * 16 + mrow) * ADIM + k + g * 16);
#pragma unroll
    for (int f = 0; f < 4; ++f) acc[f] = wmma_bf16(a.v, b[f], acc[f]);
  }

  // C layout: VGPR r, lane -> (M = r + 8*g, N = mrow + f*16)
#pragma unroll
  for (int f = 0; f < 4; ++f) {
    int n = n0 + f * 16 + mrow;
    float bn = (MODE == 0) ? bias[n] : 0.0f;
#pragma unroll
    for (int r = 0; r < 8; ++r) {
      int m = m0 + r + 8 * g;
      float x = acc[f][r];
      if (MODE == 0) outb[(size_t)m * ADIM + n] = (bf16_t)((x + bn) * scale);
      if (MODE == 1) outb[(size_t)m * ADIM + n] = (bf16_t)x;
      if (MODE == 2) outb[(size_t)n * NKS + m]  = (bf16_t)x;       // V^T store
      if (MODE == 3) outf[(size_t)m * ADIM + n] = 1.0f / (1.0f + __expf(-x));
    }
  }
}

// ---------------------------------------------------------------------------
// Flash attention with WMMA, row-per-lane softmax.
// 256 threads = 8 independent waves; wave = (head h, 16 q-rows), NK in 32-chunks.
// Per chunk: preload K/V/bias -> 4 S-wmma -> S via LDS to row-per-lane domain
// -> in-register max/sum trees (+1 xor-16 shuffle) -> exp, pack P A-frag in
// registers -> rescale O -> 4 PV-wmma (V frags loaded at chunk top).
// ---------------------------------------------------------------------------
__global__ __launch_bounds__(256) void flash_attn_wmma(
    const bf16_t* __restrict__ qp,   // [NQ][1024] bf16 (scaled, biased)
    const bf16_t* __restrict__ kp,   // [NK][1024] bf16
    const bf16_t* __restrict__ vt,   // [1024][NK] bf16 (V^T, N-major)
    const float* __restrict__ bias,  // [H][NQ][NK] f32
    const float* __restrict__ gate,  // [NQ][1024] f32
    float* __restrict__ out) {       // [NQ][1024] f32
  __shared__ __align__(64) float lds_s[8][16 * SROW];  // raw S tile per wave

  const int lane = threadIdx.x & 31;
  const int wave = threadIdx.x >> 5;
  const int mrow = lane & 15;   // C-frag N col; also this lane's owned S row
  const int g    = lane >> 4;
  const int h    = blockIdx.y;
  const int q0   = blockIdx.x * 128 + wave * 16;
  float* sbuf = lds_s[wave];

  // Q A-fragments for kd-steps 0..31 and 32..63, loaded once.
  AFrag qf[2];
  const bf16_t* qrow = qp + (size_t)(q0 + mrow) * ADIM + h * HDIM;
#pragma unroll
  for (int s = 0; s < 2; ++s) {
    qf[s].h2[0] = *(const v8bf*)(qrow + s * 32 + g * 16);
    qf[s].h2[1] = *(const v8bf*)(qrow + s * 32 + g * 16 + 8);
  }

  v8f o[4] = {};
  float rmax = -__builtin_inff();  // per-lane: stats of owned row mrow
  float rsum = 0.0f;

  // per-lane bias pointer for owned row; lane covers k = g*16+{0..15}
  const float* bro = bias + ((size_t)h * NQS + (q0 + mrow)) * NKS;
  const bf16_t* vbase = vt + (size_t)h * HDIM * NKS;

  for (int k0 = 0; k0 < NKS; k0 += 32) {
    if (k0 + 32 < NKS)  // hint next tile of the one true HBM stream (256MB)
      __builtin_prefetch(bro + k0 + 32 + g * 8, 0, 0);

    // ---- preload ALL chunk operands: 4 K-frags, 4 V-frags, 2 bias vectors ----
    v16bf kb[4];
#pragma unroll
    for (int f = 0; f < 2; ++f) {
      const bf16_t* krow = kp + (size_t)(k0 + f * 16 + mrow) * ADIM + h * HDIM;
#pragma unroll
      for (int s = 0; s < 2; ++s)
        kb[f * 2 + s] = *(const v16bf*)(krow + s * 32 + g * 16);
    }
    v16bf vb[4];
#pragma unroll
    for (int f2 = 0; f2 < 4; ++f2)
      vb[f2] = *(const v16bf*)(vbase + (size_t)(f2 * 16 + mrow) * NKS + k0 + g * 16);
    v8f b0 = *(const v8f*)(bro + k0 + g * 16);       // 32B-aligned vector loads
    v8f b1 = *(const v8f*)(bro + k0 + g * 16 + 8);

    // ---- S = Q @ K^T ----
    v8f s2[2] = {};
    s2[0] = wmma_bf16(qf[0].v, kb[0], s2[0]);
    s2[0] = wmma_bf16(qf[1].v, kb[1], s2[0]);
    s2[1] = wmma_bf16(qf[0].v, kb[2], s2[1]);
    s2[1] = wmma_bf16(qf[1].v, kb[3], s2[1]);

    // ---- raw S (C-layout) -> LDS row-major [16][SROW] ----
#pragma unroll
    for (int f = 0; f < 2; ++f)
#pragma unroll
      for (int r = 0; r < 8; ++r)
        sbuf[(r + 8 * g) * SROW + f * 16 + mrow] = s2[f][r];
    __asm__ volatile("s_wait_dscnt 0" ::: "memory");  // cross-lane LDS RAW

    // ---- lane reads its row's A-slot halves + adds bias ----
    const float* srow = sbuf + mrow * SROW;
    v8f x0 = *(const v8f*)(srow + g * 16);
    v8f x1 = *(const v8f*)(srow + g * 16 + 8);
    x0 += b0;
    x1 += b1;

    // ---- online softmax, per-lane scalars ----
    v8f mv = x0;
#pragma unroll
    for (int j = 0; j < 8; ++j) mv[j] = fmaxf(mv[j], x1[j]);
    float mx = fmaxf(fmaxf(fmaxf(mv[0], mv[1]), fmaxf(mv[2], mv[3])),
                     fmaxf(fmaxf(mv[4], mv[5]), fmaxf(mv[6], mv[7])));
    mx = fmaxf(mx, __shfl_xor(mx, 16, 32));  // combine the two row halves
    float nm   = fmaxf(rmax, mx);
    float corr = __expf(rmax - nm);
    rmax = nm;

    AFrag pa;
    float ls = 0.0f;
#pragma unroll
    for (int j = 0; j < 8; ++j) {
      float e0 = __expf(x0[j] - nm);
      float e1 = __expf(x1[j] - nm);
      ls += e0 + e1;
      pa.e[j]     = (bf16_t)e0;  // A-frag: e[j] -> k=g*16+j
      pa.e[8 + j] = (bf16_t)e1;  //           e[8+j] -> k=g*16+8+j
    }
    ls += __shfl_xor(ls, 16, 32);
    rsum = rsum * corr + ls;

    // ---- broadcast corr back to C-layout rows, rescale O ----
    float c[8];
#pragma unroll
    for (int r = 0; r < 8; ++r) c[r] = __shfl(corr, r + 8 * g, 32);
#pragma unroll
    for (int f2 = 0; f2 < 4; ++f2)
#pragma unroll
      for (int r = 0; r < 8; ++r) o[f2][r] *= c[r];

    // ---- O += P @ V (V fragments preloaded at chunk top) ----
#pragma unroll
    for (int f2 = 0; f2 < 4; ++f2) o[f2] = wmma_bf16(pa.v, vb[f2], o[f2]);
    // LDS ops are in-order per wave: next iter's stores cannot pass the loads.
  }

  // ---- finalize: 1/rowsum broadcast, gate, store f32 ----
  float inv = rsum > 0.0f ? 1.0f / rsum : 0.0f;
  float invr[8];
#pragma unroll
  for (int r = 0; r < 8; ++r) invr[r] = __shfl(inv, r + 8 * g, 32);
#pragma unroll
  for (int f2 = 0; f2 < 4; ++f2) {
    int n = h * HDIM + f2 * 16 + mrow;
#pragma unroll
    for (int r = 0; r < 8; ++r) {
      size_t idx = (size_t)(q0 + r + 8 * g) * ADIM + n;
      out[idx] = o[f2][r] * invr[r] * gate[idx];
    }
  }
}

// ---------------------------------------------------------------------------
// launch
// ---------------------------------------------------------------------------
extern "C" void kernel_launch(void* const* d_in, const int* in_sizes, int n_in,
                              void* d_out, int out_size, void* d_ws, size_t ws_size,
                              hipStream_t stream) {
  const float* q_data = (const float*)d_in[0];   // [1,2048,1024]
  const float* m_data = (const float*)d_in[1];   // [1,2048,1024]
  const float* bbias  = (const float*)d_in[2];   // [1,16,2048,2048]
  const float* qw     = (const float*)d_in[3];   // [1024,16,64]
  const float* qb     = (const float*)d_in[4];   // [1,16,64]
  const float* kw     = (const float*)d_in[5];
  const float* vw     = (const float*)d_in[6];
  const float* gw     = (const float*)d_in[7];
  float* out          = (float*)d_out;           // [1,2048,16,64]

  uint8_t* ws = (uint8_t*)d_ws;
  const size_t MB = 1u << 20;
  bf16_t* q_bf  = (bf16_t*)(ws + 0 * MB);   // 4 MiB  [2048][1024]
  bf16_t* m_bf  = (bf16_t*)(ws + 4 * MB);   // 4 MiB
  bf16_t* qwt   = (bf16_t*)(ws + 8 * MB);   // 2 MiB  [n][k]
  bf16_t* kwt   = (bf16_t*)(ws + 10 * MB);  // 2 MiB
  bf16_t* vwt   = (bf16_t*)(ws + 12 * MB);  // 2 MiB
  bf16_t* gwt   = (bf16_t*)(ws + 14 * MB);  // 2 MiB
  bf16_t* qproj = (bf16_t*)(ws + 16 * MB);  // 4 MiB  [2048][1024]
  bf16_t* kproj = (bf16_t*)(ws + 20 * MB);  // 4 MiB
  bf16_t* vtb   = (bf16_t*)(ws + 24 * MB);  // 4 MiB  [1024][2048] V^T
  float*  gatef = (float*)(ws + 28 * MB);   // 8 MiB  [2048][1024]

  const int nelem = NQS * ADIM;
  cvt_f32_to_bf16<<<nelem / 256, 256, 0, stream>>>(q_data, q_bf, nelem);
  cvt_f32_to_bf16<<<nelem / 256, 256, 0, stream>>>(m_data, m_bf, nelem);

  dim3 tb(16, 16), tg(ADIM / 16, ADIM / 16);
  transpose_to_bf16<<<tg, tb, 0, stream>>>(qw, qwt);
  transpose_to_bf16<<<tg, tb, 0, stream>>>(kw, kwt);
  transpose_to_bf16<<<tg, tb, 0, stream>>>(vw, vwt);
  transpose_to_bf16<<<tg, tb, 0, stream>>>(gw, gwt);

  const float scale = 0.125f;  // KD^-0.5, KD = 64
  dim3 gg(NQS / 64, ADIM / 64);
  proj_gemm<0><<<gg, 128, 0, stream>>>(q_bf, qwt, qb, qproj, nullptr, scale);
  proj_gemm<1><<<gg, 128, 0, stream>>>(m_bf, kwt, nullptr, kproj, nullptr, 1.0f);
  proj_gemm<2><<<gg, 128, 0, stream>>>(m_bf, vwt, nullptr, vtb, nullptr, 1.0f);
  proj_gemm<3><<<gg, 128, 0, stream>>>(q_bf, gwt, nullptr, nullptr, gatef, 1.0f);

  dim3 ag(NQS / 128, NHEAD);
  flash_attn_wmma<<<ag, 256, 0, stream>>>(qproj, kproj, vtb, bbias, gatef, out);
}
